// MultiHeadSelfAttention_4621384810943
// MI455X (gfx1250) — matrix operand order repaired and checked by reading, not compiled, not run on hardware
//
#include <hip/hip_runtime.h>

// ---------------------------------------------------------------------------
// MultiHeadSelfAttention forward for MI455X (gfx1250, wave32, WMMA).
// All matmuls use v_wmma_f32_16x16x32_f16 (f16 A/B, f32 accumulate).
// Each wave computes a 16x64 output tile: 4 independent accumulator chains
// (ILP across the XDL pipe) and 4x reuse of each A fragment.
// d_out = [ output (2*2048*1024 f32) | attn_weights (2*16*2048*2048 f32) ]
// ---------------------------------------------------------------------------

typedef __attribute__((ext_vector_type(16))) _Float16 v16h;
typedef __attribute__((ext_vector_type(8)))  _Float16 v8h;
typedef __attribute__((ext_vector_type(4)))  _Float16 v4h;
typedef __attribute__((ext_vector_type(8)))  float    v8f;
typedef __attribute__((ext_vector_type(4)))  float    v4f;

union V16H { v16h v; v8h h8[2]; };
union V8H4 { v8h v; v4h h4[2]; };

#define B_   2
#define S_   2048
#define D_   1024
#define H_   16
#define DH_  64
#define NB_  4      // N-tiles (16 wide each) per wave -> 16x64 wave tile

// ---- A fragment (16x32 f16, MxK). Per CDNA5 ISA 7.12.2:
// lanes 0-15: m=lane, elems 0..7 -> K 0..7, elems 8..15 -> K 16..23
// lanes 16-31: m=lane-16, elems 0..7 -> K 8..15, elems 8..15 -> K 24..31
__device__ __forceinline__ v16h load_a_f16(const _Float16* a, int lda, int lane) {
    int m  = lane & 15;
    int kb = (lane < 16) ? 0 : 8;
    const _Float16* p = a + (size_t)m * lda + kb;
    V16H r;
    r.h8[0] = *(const v8h*)(p);
    r.h8[1] = *(const v8h*)(p + 16);
    return r.v;
}

// Same fragment but A is f32 in memory (attention probabilities): convert on load.
__device__ __forceinline__ v16h load_a_f32(const float* a, int lda, int lane) {
    int m  = lane & 15;
    int kb = (lane < 16) ? 0 : 8;
    const float* p = a + (size_t)m * lda + kb;
    v4f f0 = *(const v4f*)(p);
    v4f f1 = *(const v4f*)(p + 4);
    v4f f2 = *(const v4f*)(p + 16);
    v4f f3 = *(const v4f*)(p + 20);
    V8H4 u0, u1;
    u0.h4[0] = __builtin_convertvector(f0, v4h);
    u0.h4[1] = __builtin_convertvector(f1, v4h);
    u1.h4[0] = __builtin_convertvector(f2, v4h);
    u1.h4[1] = __builtin_convertvector(f3, v4h);
    V16H r;
    r.h8[0] = u0.v;
    r.h8[1] = u1.v;
    return r.v;
}

// ---- B fragment (32x16 f16, KxN), supplied as Bt[n][k] (row-major B^T):
// lanes 0-15: n=lane, K 0..7,16..23 ; lanes 16-31: n=lane-16, K 8..15,24..31 (same as A)
__device__ __forceinline__ v16h load_b_f16(const _Float16* bt, int ldb, int lane) {
    int n  = lane & 15;
    int kb = (lane < 16) ? 0 : 8;
    const _Float16* p = bt + (size_t)n * ldb + kb;
    V16H r;
    r.h8[0] = *(const v8h*)(p);
    r.h8[1] = *(const v8h*)(p + 16);
    return r.v;
}

// ---------------------------------------------------------------------------
// Generic WMMA GEMM: C[z] = scale * (A[z] @ Bt[z]^T) + bias
//   AMODE: 0 = A is f16, 1 = A is f32 (converted per fragment)
//   OMODE: 0 = f32 out, 1 = f16 out, 2 = f16 out transposed per batch (V^T)
// One wave computes a 16x64 tile (4 accumulators); 8 waves per block;
// grid.z = batch*head.  Requires N % 64 == 0 and (M/16)*(N/64) % 8 == 0.
// ---------------------------------------------------------------------------
template<int AMODE, int OMODE>
__global__ __launch_bounds__(256) void gemm_wmma_kernel(
    const void* __restrict__ Av, int lda, long long aSB, long long aSH,
    const _Float16* __restrict__ Btv, int ldb, long long bSB, long long bSH,
    void* __restrict__ Cv, int ldc, long long cSB, long long cSH,
    const float* __restrict__ bias, float scale,
    int M, int N, int Kd, int Hdiv, int Sdim)
{
    const int lane   = threadIdx.x & 31;
    const int wave   = threadIdx.x >> 5;
    const int tilesN = N >> 6;                      // 64-wide wave tiles
    const int tilesM = M >> 4;
    const int tile   = blockIdx.x * 8 + wave;
    if (tile >= tilesM * tilesN) return;            // wave-uniform
    const int mt = tile / tilesN;
    const int nt = tile - mt * tilesN;
    const int z  = blockIdx.z;
    const int bz = z / Hdiv;
    const int hz = z - bz * Hdiv;

    const _Float16* Bt = Btv + bz * bSB + hz * bSH + (size_t)(nt * 64) * ldb;

    v8f acc0 = {}, acc1 = {}, acc2 = {}, acc3 = {};
    if (AMODE == 0) {
        const _Float16* A = (const _Float16*)Av + bz * aSB + hz * aSH
                          + (size_t)(mt * 16) * lda;
        for (int k0 = 0; k0 < Kd; k0 += 32) {
            v16h af = load_a_f16(A + k0, lda, lane);
            v16h b0 = load_b_f16(Bt + k0,                       ldb, lane);
            v16h b1 = load_b_f16(Bt + k0 + (size_t)16 * ldb,    ldb, lane);
            v16h b2 = load_b_f16(Bt + k0 + (size_t)32 * ldb,    ldb, lane);
            v16h b3 = load_b_f16(Bt + k0 + (size_t)48 * ldb,    ldb, lane);
            acc0 = __builtin_amdgcn_wmma_f32_16x16x32_f16(false, af, false, b0, (short)0, acc0, false, false);
            acc1 = __builtin_amdgcn_wmma_f32_16x16x32_f16(false, af, false, b1, (short)0, acc1, false, false);
            acc2 = __builtin_amdgcn_wmma_f32_16x16x32_f16(false, af, false, b2, (short)0, acc2, false, false);
            acc3 = __builtin_amdgcn_wmma_f32_16x16x32_f16(false, af, false, b3, (short)0, acc3, false, false);
        }
    } else {
        const float* A = (const float*)Av + bz * aSB + hz * aSH
                       + (size_t)(mt * 16) * lda;
        for (int k0 = 0; k0 < Kd; k0 += 32) {
            v16h af = load_a_f32(A + k0, lda, lane);
            v16h b0 = load_b_f16(Bt + k0,                       ldb, lane);
            v16h b1 = load_b_f16(Bt + k0 + (size_t)16 * ldb,    ldb, lane);
            v16h b2 = load_b_f16(Bt + k0 + (size_t)32 * ldb,    ldb, lane);
            v16h b3 = load_b_f16(Bt + k0 + (size_t)48 * ldb,    ldb, lane);
            acc0 = __builtin_amdgcn_wmma_f32_16x16x32_f16(false, af, false, b0, (short)0, acc0, false, false);
            acc1 = __builtin_amdgcn_wmma_f32_16x16x32_f16(false, af, false, b1, (short)0, acc1, false, false);
            acc2 = __builtin_amdgcn_wmma_f32_16x16x32_f16(false, af, false, b2, (short)0, acc2, false, false);
            acc3 = __builtin_amdgcn_wmma_f32_16x16x32_f16(false, af, false, b3, (short)0, acc3, false, false);
        }
    }

    // C/D layout: lane n = lane&15 ; VGPR r -> m = r + 8*(lane>=16)
    const int nl = lane & 15;
    const int mb = (lane >> 4) * 8;
    v8f accs[NB_] = {acc0, acc1, acc2, acc3};

#pragma unroll
    for (int j = 0; j < NB_; ++j) {
        const int ng = nt * 64 + j * 16 + nl;
        const float bn = (bias != nullptr) ? bias[ng] : 0.0f;
        if (OMODE == 0) {
            float* C = (float*)Cv + bz * cSB + hz * cSH;
#pragma unroll
            for (int r = 0; r < 8; ++r) {
                int mg = mt * 16 + mb + r;
                C[(size_t)mg * ldc + ng] = accs[j][r] * scale + bn;
            }
        } else if (OMODE == 1) {
            _Float16* C = (_Float16*)Cv + bz * cSB + hz * cSH;
#pragma unroll
            for (int r = 0; r < 8; ++r) {
                int mg = mt * 16 + mb + r;
                C[(size_t)mg * ldc + ng] = (_Float16)(accs[j][r] * scale + bn);
            }
        } else {
            // transposed per batch: C[b][n][s], m = b*Sdim + s
            _Float16* C = (_Float16*)Cv;
#pragma unroll
            for (int r = 0; r < 8; ++r) {
                int mg = mt * 16 + mb + r;
                int bb = mg / Sdim;
                int s  = mg - bb * Sdim;
                C[(size_t)bb * cSB + (size_t)ng * ldc + s] =
                    (_Float16)(accs[j][r] * scale + bn);
            }
        }
    }
}

// ---------------------------------------------------------------------------
// f32 -> f16 elementwise (vectorized x4)
// ---------------------------------------------------------------------------
__global__ __launch_bounds__(256) void cvt_f16x4_kernel(
    const float* __restrict__ in, _Float16* __restrict__ out, int n4)
{
    int i = blockIdx.x * 256 + threadIdx.x;
    if (i < n4) {
        v4f f = ((const v4f*)in)[i];
        ((v4h*)out)[i] = __builtin_convertvector(f, v4h);
    }
}

// ---------------------------------------------------------------------------
// W (KxN f32) -> W^T (NxK f16), LDS-tiled transpose. block (32,8), 32x32 tiles.
// ---------------------------------------------------------------------------
__global__ __launch_bounds__(256) void transpose_cvt_kernel(
    const float* __restrict__ in, _Float16* __restrict__ out, int K, int N)
{
    __shared__ _Float16 tile[32][33];
    int k0 = blockIdx.y * 32;
    int n0 = blockIdx.x * 32;
    int tx = threadIdx.x;
    for (int i = threadIdx.y; i < 32; i += 8)
        tile[i][tx] = (_Float16)in[(size_t)(k0 + i) * N + n0 + tx];
    __syncthreads();
    for (int i = threadIdx.y; i < 32; i += 8)
        out[(size_t)(n0 + i) * K + k0 + tx] = tile[tx][i];
}

// ---------------------------------------------------------------------------
// Masked softmax in-place over rows of attn (row length S_=2048).
// One 256-thread block per row; 8 columns per thread; LDS + shuffle reductions.
// ---------------------------------------------------------------------------
__global__ __launch_bounds__(256) void softmax_kernel(
    float* __restrict__ attn, const int* __restrict__ mask, int HS)
{
    const int row = blockIdx.x;
    float* p = attn + (size_t)row * S_;
    const int* mrow = mask + (size_t)(row / HS) * S_;
    const int t = threadIdx.x;
    const int lane = t & 31, wave = t >> 5;
    __shared__ float red[8];

    float vals[8];
    float mx = -INFINITY;
#pragma unroll
    for (int j = 0; j < 8; ++j) {
        int c = t + j * 256;
        float v = p[c];
        vals[j] = (mrow[c] == 0) ? -INFINITY : v;
        mx = fmaxf(mx, vals[j]);
    }
#pragma unroll
    for (int off = 16; off > 0; off >>= 1)
        mx = fmaxf(mx, __shfl_xor(mx, off, 32));
    if (lane == 0) red[wave] = mx;
    __syncthreads();
#pragma unroll
    for (int w = 0; w < 8; ++w) mx = fmaxf(mx, red[w]);

    float sum = 0.0f;
#pragma unroll
    for (int j = 0; j < 8; ++j) {
        vals[j] = __expf(vals[j] - mx);
        sum += vals[j];
    }
#pragma unroll
    for (int off = 16; off > 0; off >>= 1)
        sum += __shfl_xor(sum, off, 32);
    __syncthreads();
    if (lane == 0) red[wave] = sum;
    __syncthreads();
    sum = 0.0f;
#pragma unroll
    for (int w = 0; w < 8; ++w) sum += red[w];

    float inv = 1.0f / sum;
#pragma unroll
    for (int j = 0; j < 8; ++j)
        p[t + j * 256] = vals[j] * inv;
}

// ---------------------------------------------------------------------------
extern "C" void kernel_launch(void* const* d_in, const int* in_sizes, int n_in,
                              void* d_out, int out_size, void* d_ws, size_t ws_size,
                              hipStream_t stream)
{
    const float* x     = (const float*)d_in[0];
    const int*   amask = (const int*)  d_in[1];
    const float* Wq    = (const float*)d_in[2];
    const float* bq    = (const float*)d_in[3];
    const float* Wk    = (const float*)d_in[4];
    const float* bk    = (const float*)d_in[5];
    const float* Wv    = (const float*)d_in[6];
    const float* bv    = (const float*)d_in[7];
    const float* Wo    = (const float*)d_in[8];
    const float* bo    = (const float*)d_in[9];

    float* out  = (float*)d_out;                       // (B,S,D)
    float* attn = out + (size_t)B_ * S_ * D_;          // (B,H,S,S)

    // workspace layout (f16), 48 MB total
    char* ws = (char*)d_ws;
    _Float16* xh  = (_Float16*)(ws);                         // 8 MB  (B*S, D)
    _Float16* wqT = (_Float16*)(ws + (8u  << 20));           // 2 MB  (N,K)
    _Float16* wkT = (_Float16*)(ws + (10u << 20));
    _Float16* wvT = (_Float16*)(ws + (12u << 20));
    _Float16* woT = (_Float16*)(ws + (14u << 20));
    _Float16* Qh  = (_Float16*)(ws + (16u << 20));           // 8 MB  (B*S, D)
    _Float16* Kh  = (_Float16*)(ws + (24u << 20));           // 8 MB  (B*S, D)
    _Float16* Vth = (_Float16*)(ws + (32u << 20));           // 8 MB  (B, D, S)
    _Float16* AOh = (_Float16*)(ws + (40u << 20));           // 8 MB  (B*S, D)

    const long long SD  = (long long)S_ * D_;                // 2097152
    const long long SS  = (long long)S_ * S_;                // 4194304
    const long long HSS = (long long)H_ * SS;                // 67108864

    // 1) x -> f16
    cvt_f16x4_kernel<<<(B_ * S_ * D_ / 4 + 255) / 256, 256, 0, stream>>>(
        x, xh, B_ * S_ * D_ / 4);

    // 2) weight transposes (f32 KxN -> f16 NxK)
    dim3 tb(32, 8), tg(D_ / 32, D_ / 32);
    transpose_cvt_kernel<<<tg, tb, 0, stream>>>(Wq, wqT, D_, D_);
    transpose_cvt_kernel<<<tg, tb, 0, stream>>>(Wk, wkT, D_, D_);
    transpose_cvt_kernel<<<tg, tb, 0, stream>>>(Wv, wvT, D_, D_);
    transpose_cvt_kernel<<<tg, tb, 0, stream>>>(Wo, woT, D_, D_);

    // 3) QKV projections: (4096x1024) @ (1024x1024) + bias
    {
        dim3 g((4096 / 16) * (1024 / 64) / 8, 1, 1);   // 512 blocks
        gemm_wmma_kernel<0, 1><<<g, 256, 0, stream>>>(
            xh, D_, 0, 0, wqT, D_, 0, 0, Qh, D_, 0, 0,
            bq, 1.0f, B_ * S_, D_, D_, 1, 0);
        gemm_wmma_kernel<0, 1><<<g, 256, 0, stream>>>(
            xh, D_, 0, 0, wkT, D_, 0, 0, Kh, D_, 0, 0,
            bk, 1.0f, B_ * S_, D_, D_, 1, 0);
        // V written transposed per batch: Vth[b][d][s]
        gemm_wmma_kernel<0, 2><<<g, 256, 0, stream>>>(
            xh, D_, 0, 0, wvT, D_, 0, 0, Vth, S_, SD, 0,
            bv, 1.0f, B_ * S_, D_, D_, 1, S_);
    }

    // 4) scores = (Q_h @ K_h^T) / sqrt(64) -> attn region of d_out (f32)
    {
        dim3 g((S_ / 16) * (S_ / 64) / 8, 1, B_ * H_); // 512 x 1 x 32
        gemm_wmma_kernel<0, 0><<<g, 256, 0, stream>>>(
            Qh, D_, SD, DH_,        // A: Q[b][q][h*64 + k]
            Kh, D_, SD, DH_,        // Bt: K[b][kk][h*64 + k]
            attn, S_, HSS, SS,      // C: attn[b][h][q][kk]
            nullptr, 0.125f, S_, S_, DH_, H_, 0);
    }

    // 5) softmax with mask, in place on attn
    softmax_kernel<<<B_ * H_ * S_, 256, 0, stream>>>(attn, amask, H_ * S_);

    // 6) attn_out = attn @ V  (A f32 -> f16 on load), out f16 (B,S,H*Dh)
    {
        dim3 g((S_ / 16) * (DH_ / 64) / 8, 1, B_ * H_); // 16 x 1 x 32
        gemm_wmma_kernel<1, 1><<<g, 256, 0, stream>>>(
            attn, S_, HSS, SS,                       // A: attn[b][h][q][kk]
            Vth, S_, SD, (long long)DH_ * S_,        // Bt: Vth[b][h*64+d][kk]
            AOh, D_, SD, DH_,                        // C: AOh[b][q][h*64+d]
            nullptr, 1.0f, S_, DH_, S_, H_, 0);
    }

    // 7) output = attn_out @ Wo + bo  (f32 to d_out)
    {
        dim3 g((4096 / 16) * (1024 / 64) / 8, 1, 1);   // 512 blocks
        gemm_wmma_kernel<0, 0><<<g, 256, 0, stream>>>(
            AOh, D_, 0, 0, woT, D_, 0, 0, out, D_, 0, 0,
            bo, 1.0f, B_ * S_, D_, D_, 1, 0);
    }
}
